// InvariantPointAttention_71064528879835
// MI455X (gfx1250) — matrix operand order repaired and checked by reading, not compiled, not run on hardware
//
#include <hip/hip_runtime.h>
#include <hip/hip_bf16.h>

// AF2 IPA dims
#define NRES 768
#define CSD  384
#define CZD  128
#define NHD  12
#define HPAD 16                        /* heads padded to 16 for guard-free loads */
#define N2   (768 * 768)

#define SCALE_SINGLE 0.25f             /* 1/sqrt(16) */
#define SCALE_FRAME  (-0.11785113f)    /* -1/sqrt(72) */

typedef __attribute__((ext_vector_type(16))) _Float16 v16h;
typedef __attribute__((ext_vector_type(8)))  float    v8f;

// ---------------------------------------------------------------------------
// Fragment layouts (wave32, v_wmma_f32_16x16x32_f16 per CDNA5 ISA 7.12.2):
//  A (16x32 MxK): lane&15 = M; elements e<8 -> K = 8*(lane>>4)+e,
//                 e>=8 -> K = 16 + 8*(lane>>4) + (e-8)  -> two contiguous runs.
//  B (32x16 KxN): lane&15 = N; ISA uses A's order; this code loads A and B as K = 16*(lane>>4) + e.
//  C/D (16x16):   reg r -> M = 8*(lane>>4)+r, N = lane&15.
// ---------------------------------------------------------------------------
__device__ __forceinline__ v16h pack16(float4 x0, float4 x1, float4 x2, float4 x3) {
  v16h f;
  f[0]  = (_Float16)x0.x; f[1]  = (_Float16)x0.y; f[2]  = (_Float16)x0.z; f[3]  = (_Float16)x0.w;
  f[4]  = (_Float16)x1.x; f[5]  = (_Float16)x1.y; f[6]  = (_Float16)x1.z; f[7]  = (_Float16)x1.w;
  f[8]  = (_Float16)x2.x; f[9]  = (_Float16)x2.y; f[10] = (_Float16)x2.z; f[11] = (_Float16)x2.w;
  f[12] = (_Float16)x3.x; f[13] = (_Float16)x3.y; f[14] = (_Float16)x3.z; f[15] = (_Float16)x3.w;
  return f;
}

// unconditional vectorized A fragment: ap points at row (m0+lm), col k0
__device__ __forceinline__ v16h load_a_frag(const float* __restrict__ ap, int lh) {
  const float* p = ap + lh * 16;
  return pack16(*(const float4*)(p + 0),  *(const float4*)(p + 4),
                *(const float4*)(p + 8), *(const float4*)(p + 12));
}

// ---------------------------------------------------------------------------
// Generic strided-batched GEMM, f32 io, f16 WMMA accumulate-f32.
// Loads are UNGUARDED: caller guarantees ceil16(M) rows of A and ceil16(Nc)
// cols of B are readable (buffers padded). Stores guarded by Mstore/Nstore.
// btrans: B stored transposed [Nc][K] (contiguous-K fast path).
// transC: store C[n*ldc+m].
// ---------------------------------------------------------------------------
__global__ void gemm_f16_wmma(const float* __restrict__ A, const float* __restrict__ B,
                              float* __restrict__ C, const float* __restrict__ bias,
                              int M, int Nc, int K, int Mstore, int Nstore,
                              long lda, long ldb, long ldc,
                              long sA, long sB, long sC, int btrans, int transC) {
  const int lane = threadIdx.x & 31;
  const int wid  = (blockIdx.x * blockDim.x + threadIdx.x) >> 5;
  const int tilesN = Nc >> 4, tilesM = M >> 4;
  if (wid >= tilesM * tilesN) return;             // wave-uniform
  const int mt = wid / tilesN, nt = wid % tilesN;
  const int b = blockIdx.y;
  A += (size_t)b * sA; B += (size_t)b * sB; C += (size_t)b * sC;
  const int m0 = mt * 16, n0 = nt * 16;
  const int lm = lane & 15, lh = lane >> 4;

  const float* arow = A + (size_t)(m0 + lm) * lda;
  v8f acc = {};
  for (int k0 = 0; k0 < K; k0 += 32) {
    if (k0 + 64 <= K) {                           // scalar condition: no EXEC churn
      __builtin_prefetch(arow + k0 + 32 + lh * 8, 0, 1);
      if (!btrans)
        __builtin_prefetch(B + (size_t)(k0 + 32 + lh * 16) * ldb + n0 + lm, 0, 1);
    }
    const v16h af = load_a_frag(arow + k0, lh);
    v16h bf;
    if (btrans) {
      const float* bp = B + (size_t)(n0 + lm) * ldb + k0 + lh * 16;
      bf = pack16(*(const float4*)(bp + 0), *(const float4*)(bp + 4),
                  *(const float4*)(bp + 8), *(const float4*)(bp + 12));
    } else {
      const float* bp = B + (size_t)(k0 + lh * 16) * ldb + n0 + lm;
#pragma unroll
      for (int e = 0; e < 16; ++e) bf[e] = (_Float16)bp[(size_t)e * ldb];
    }
    acc = __builtin_amdgcn_wmma_f32_16x16x32_f16(false, af, false, bf,
                                                 (short)0, acc, false, false);
  }
#pragma unroll
  for (int r = 0; r < 8; ++r) {
    const int m = m0 + lh * 8 + r;
    const int n = n0 + lm;
    if (m < Mstore && n < Nstore) {
      const float val = acc[r] + (bias ? bias[n] : 0.0f);
      if (transC) C[(size_t)n * ldc + m] = val;
      else        C[(size_t)m * ldc + n] = val;
    }
  }
}

// ---------------------------------------------------------------------------
// o_z with CDNA5 async Global->LDS staging: one workgroup per residue i.
// out[h, c] = sum_j a[h, i, j] * z[i, j, c]  (heads padded to 16, 8 N-tiles)
// z[i] is streamed through LDS in double-buffered 32x128 f32 slabs using
// GLOBAL_LOAD_ASYNC_TO_LDS_B128 (ASYNCcnt), B fragments served by ds_load.
// ---------------------------------------------------------------------------
__global__ void __launch_bounds__(256)
oz_async_wmma(const float* __restrict__ a, const float* __restrict__ z,
              float* __restrict__ att) {
  __shared__ float smem[2][32 * CZD];              // 2 x 16 KB slabs
  const int i    = blockIdx.x;
  const int tid  = threadIdx.x;
  const int lane = tid & 31;
  const int wv   = tid >> 5;                       // wave -> N-tile (8 tiles of 16)
  const int lm   = lane & 15, lh = lane >> 4;
  const float* zb = z + (size_t)i * NRES * CZD;

  // issue one 32x128 f32 slab (1024 float4s; 4 async b128 per thread)
  auto issue = [&](int buf, int j0) {
    const float* src = zb + (size_t)j0 * CZD;
#pragma unroll
    for (int c = 0; c < 4; ++c) {
      const int idx = c * 256 + tid;               // float4 index within slab
      unsigned lds_off = (unsigned)(size_t)&smem[buf][idx * 4];
      unsigned long long ga = (unsigned long long)(size_t)(src + idx * 4);
      asm volatile("global_load_async_to_lds_b128 %0, %1, off"
                   :: "v"(lds_off), "v"(ga) : "memory");
    }
  };

  const float* arow = a + (size_t)lm * N2 + (size_t)i * NRES;  // a[h=lm][i][j]

  issue(0, 0);
  v8f acc = {};
  for (int k0 = 0; k0 < NRES; k0 += 32) {
    const int buf = (k0 >> 5) & 1;
    asm volatile("s_wait_asynccnt 0x0" ::: "memory");  // own slab loads landed
    __syncthreads();                                   // everyone's landed
    if (k0 + 32 < NRES) issue(buf ^ 1, k0 + 32);       // prefetch next slab

    const v16h af = load_a_frag(arow + k0, lh);
    v16h bf;
    const float* sp = &smem[buf][(lh * 16) * CZD + wv * 16 + lm];
#pragma unroll
    for (int e = 0; e < 16; ++e) bf[e] = (_Float16)sp[e * CZD];
    acc = __builtin_amdgcn_wmma_f32_16x16x32_f16(false, af, false, bf,
                                                 (short)0, acc, false, false);
    // ds reads complete before each wave reaches the next barrier (WMMA dep)
  }
#pragma unroll
  for (int r = 0; r < 8; ++r) {
    const int m = lh * 8 + r;                      // head
    const int n = wv * 16 + lm;                    // channel
    if (m < NHD) att[(size_t)i * 2112 + 192 + (size_t)m * CZD + n] = acc[r];
  }
}

// ---------------------------------------------------------------------------
// Pad Wb [128,12] -> [128,16] with zero columns (keeps all B loads in-bounds
// and makes the padded logit head-rows exactly zero).
// ---------------------------------------------------------------------------
__global__ void pad_wb(const float* __restrict__ Wb, float* __restrict__ WbPad) {
  const int t = blockIdx.x * blockDim.x + threadIdx.x;
  if (t >= CZD * HPAD) return;
  const int k = t / HPAD, n = t % HPAD;
  WbPad[t] = (n < NHD) ? Wb[k * NHD + n] : 0.0f;
}

// ---------------------------------------------------------------------------
// Apply frames: g = R_n * p + t_n for qp (48 pts), kp (48), vp (96) per residue
// ---------------------------------------------------------------------------
__global__ void apply_frames(const float* __restrict__ qp, const float* __restrict__ kp,
                             const float* __restrict__ vp, const float* __restrict__ R,
                             const float* __restrict__ tr,
                             float* __restrict__ gq, float* __restrict__ gk,
                             float* __restrict__ gv) {
  const int t = blockIdx.x * blockDim.x + threadIdx.x;
  if (t >= NRES * 192) return;
  const int n = t / 192, slot = t % 192;
  const float* src; float* dst; int off;
  if (slot < 48)      { src = qp; dst = gq; off = n * 144 + slot * 3; }
  else if (slot < 96) { src = kp; dst = gk; off = n * 144 + (slot - 48) * 3; }
  else                { src = vp; dst = gv; off = n * 288 + (slot - 96) * 3; }
  const float x = src[off], y = src[off + 1], z = src[off + 2];
  const float* Rn = R + n * 9;
  const float* tn = tr + n * 3;
  dst[off + 0] = Rn[0] * x + Rn[1] * y + Rn[2] * z + tn[0];
  dst[off + 1] = Rn[3] * x + Rn[4] * y + Rn[5] * z + tn[1];
  dst[off + 2] = Rn[6] * x + Rn[7] * y + Rn[8] * z + tn[2];
}

// ---------------------------------------------------------------------------
// Pack per-head K=32 logit operands. A = [0.25*q(16) | -2*gamma*gq(12) | 0(4)]
// B = [k(16) | gk(12) | 0(4)].  Also |gq|^2, |gk|^2 sums per (h,n).
// ---------------------------------------------------------------------------
__global__ void pack_ab(const float* __restrict__ q, const float* __restrict__ k,
                        const float* __restrict__ gq, const float* __restrict__ gk,
                        const float* __restrict__ scale_head,
                        float* __restrict__ Apack, float* __restrict__ Bpack,
                        float* __restrict__ sqq, float* __restrict__ sqk) {
  const int t = blockIdx.x * blockDim.x + threadIdx.x;
  if (t >= NHD * NRES) return;
  const int h = t / NRES, n = t % NRES;
  const float gamma = SCALE_FRAME * log1pf(expf(scale_head[h]));
  float* Ao = Apack + (size_t)t * 32;
  float* Bo = Bpack + (size_t)t * 32;
#pragma unroll
  for (int e = 0; e < 16; ++e) {
    Ao[e] = SCALE_SINGLE * q[(size_t)n * 192 + h * 16 + e];
    Bo[e] = k[(size_t)n * 192 + h * 16 + e];
  }
  float sq = 0.0f, sk = 0.0f;
#pragma unroll
  for (int u = 0; u < 12; ++u) {
    const float a = gq[(size_t)n * 144 + h * 12 + u];
    const float b = gk[(size_t)n * 144 + h * 12 + u];
    sq += a * a; sk += b * b;
    Ao[16 + u] = -2.0f * gamma * a;
    Bo[16 + u] = b;
  }
#pragma unroll
  for (int u = 28; u < 32; ++u) { Ao[u] = 0.0f; Bo[u] = 0.0f; }
  sqq[t] = sq;
  sqk[t] = sk;
}

// ---------------------------------------------------------------------------
// Logit tiles: D = Apack*Bpack^T + (w_pair + gamma*(|gq_i|^2 + |gk_j|^2))
// One 16x16x32 WMMA per tile; all fragment loads are contiguous b128.
// ---------------------------------------------------------------------------
__global__ void logits_wmma(const float* __restrict__ Apack, const float* __restrict__ Bpack,
                            const float* __restrict__ sqq, const float* __restrict__ sqk,
                            const float* __restrict__ scale_head,
                            float* __restrict__ logits) {
  const int lane = threadIdx.x & 31;
  const int wid  = (blockIdx.x * blockDim.x + threadIdx.x) >> 5;
  const int tiles = NRES / 16;                               // 48
  if (wid >= tiles * tiles) return;
  const int h  = blockIdx.y;
  const int i0 = (wid / tiles) * 16, j0 = (wid % tiles) * 16;
  const int lm = lane & 15, lh = lane >> 4;
  const float gamma = SCALE_FRAME * log1pf(expf(scale_head[h]));

  const float* Ap = Apack + ((size_t)h * NRES + i0 + lm) * 32;   // A row, K=0..31
  const float* Bp = Bpack + ((size_t)h * NRES + j0 + lm) * 32 + lh * 16;
  const v16h af = load_a_frag(Ap, lh);
  const v16h bf = pack16(*(const float4*)(Bp + 0), *(const float4*)(Bp + 4),
                         *(const float4*)(Bp + 8), *(const float4*)(Bp + 12));
  v8f c;
#pragma unroll
  for (int r = 0; r < 8; ++r) {
    const int m = i0 + lh * 8 + r, n = j0 + lm;
    c[r] = logits[(size_t)h * N2 + (size_t)m * NRES + n]
         + gamma * (sqq[h * NRES + m] + sqk[h * NRES + n]);
  }
  const v8f d = __builtin_amdgcn_wmma_f32_16x16x32_f16(false, af, false, bf,
                                                       (short)0, c, false, false);
#pragma unroll
  for (int r = 0; r < 8; ++r) {
    const int m = i0 + lh * 8 + r, n = j0 + lm;
    logits[(size_t)h * N2 + (size_t)m * NRES + n] = d[r];
  }
}

// ---------------------------------------------------------------------------
// Row softmax over j, in place. One block per (h,i) row of 768.
// ---------------------------------------------------------------------------
__global__ void softmax_rows(float* __restrict__ logits) {
  __shared__ float red[256];
  float* p = logits + (size_t)blockIdx.x * NRES;
  const int tid = threadIdx.x;
  float m = -1e30f;
  for (int j = tid; j < NRES; j += 256) m = fmaxf(m, p[j]);
  red[tid] = m; __syncthreads();
  for (int s = 128; s > 0; s >>= 1) { if (tid < s) red[tid] = fmaxf(red[tid], red[tid + s]); __syncthreads(); }
  m = red[0]; __syncthreads();
  float sum = 0.0f;
  for (int j = tid; j < NRES; j += 256) { const float e = expf(p[j] - m); p[j] = e; sum += e; }
  red[tid] = sum; __syncthreads();
  for (int s = 128; s > 0; s >>= 1) { if (tid < s) red[tid] += red[tid + s]; __syncthreads(); }
  const float inv = 1.0f / red[0];
  for (int j = tid; j < NRES; j += 256) p[j] *= inv;
}

// ---------------------------------------------------------------------------
// Pack per-head value matrix, TRANSPOSED: vgt[h][col][j], col in [v(16)|gv(24)|0(8)]
// ---------------------------------------------------------------------------
__global__ void pack_vg(const float* __restrict__ v, const float* __restrict__ gv,
                        float* __restrict__ vgt) {
  const int t = blockIdx.x * blockDim.x + threadIdx.x;
  if (t >= NHD * NRES) return;
  const int h = t / NRES, j = t % NRES;
#pragma unroll
  for (int e = 0; e < 16; ++e)
    vgt[((size_t)h * 48 + e) * NRES + j] = v[(size_t)j * 192 + h * 16 + e];
#pragma unroll
  for (int u = 0; u < 24; ++u)
    vgt[((size_t)h * 48 + 16 + u) * NRES + j] = gv[(size_t)j * 288 + h * 24 + u];
#pragma unroll
  for (int u = 40; u < 48; ++u)
    vgt[((size_t)h * 48 + u) * NRES + j] = 0.0f;
}

// ---------------------------------------------------------------------------
// Finalize: copy o_s into att, inverse-frame the attended points, pts + norms.
// ---------------------------------------------------------------------------
__global__ void finalize_att(const float* __restrict__ ocat, const float* __restrict__ R,
                             const float* __restrict__ tr, float* __restrict__ att) {
  const int t = blockIdx.x * blockDim.x + threadIdx.x;
  if (t >= NHD * NRES) return;
  const int h = t / NRES, i = t % NRES;
  const float* oc = ocat + ((size_t)h * NRES + i) * 48;
  float* arow = att + (size_t)i * 2112;
#pragma unroll
  for (int e = 0; e < 16; ++e) arow[h * 16 + e] = oc[e];
  const float* Rn = R + i * 9;
  const float* tn = tr + i * 3;
#pragma unroll
  for (int p = 0; p < 8; ++p) {
    const float dx = oc[16 + p * 3 + 0] - tn[0];
    const float dy = oc[16 + p * 3 + 1] - tn[1];
    const float dz = oc[16 + p * 3 + 2] - tn[2];
    const float lx = Rn[0] * dx + Rn[3] * dy + Rn[6] * dz;   // R^T * d
    const float ly = Rn[1] * dx + Rn[4] * dy + Rn[7] * dz;
    const float lz = Rn[2] * dx + Rn[5] * dy + Rn[8] * dz;
    arow[1728 + p * 36 + h * 3 + 0] = lx;
    arow[1728 + p * 36 + h * 3 + 1] = ly;
    arow[1728 + p * 36 + h * 3 + 2] = lz;
    arow[2016 + p * 12 + h] = sqrtf(lx * lx + ly * ly + lz * lz);
  }
}

// ---------------------------------------------------------------------------
// Host orchestration
// ---------------------------------------------------------------------------
static void launch_gemm(hipStream_t st, const float* A, const float* B, float* C,
                        const float* bias, int M, int Nc, int K, int Mstore, int Nstore,
                        long lda, long ldb, long ldc,
                        int batch, long sA, long sB, long sC, int btrans, int transC) {
  const int tiles = (M / 16) * (Nc / 16);
  const int blocks = (tiles + 7) / 8;
  dim3 grid(blocks, batch);
  gemm_f16_wmma<<<grid, 256, 0, st>>>(A, B, C, bias, M, Nc, K, Mstore, Nstore,
                                      lda, ldb, ldc, sA, sB, sC, btrans, transC);
}

extern "C" void kernel_launch(void* const* d_in, const int* in_sizes, int n_in,
                              void* d_out, int out_size, void* d_ws, size_t ws_size,
                              hipStream_t stream) {
  const float* s    = (const float*)d_in[0];   // [768,384]
  const float* z    = (const float*)d_in[1];   // [768,768,128]
  const float* R    = (const float*)d_in[2];   // [768,3,3]
  const float* tr   = (const float*)d_in[3];   // [768,3]
  const float* Wq   = (const float*)d_in[4];   // [384,192]
  const float* Wk   = (const float*)d_in[5];
  const float* Wv   = (const float*)d_in[6];
  const float* Wqp  = (const float*)d_in[7];   // [384,144]
  const float* Wkp  = (const float*)d_in[8];
  const float* Wvp  = (const float*)d_in[9];   // [384,288]
  const float* Wb   = (const float*)d_in[10];  // [128,12]
  const float* Wout = (const float*)d_in[11];  // [2112,384]
  const float* bout = (const float*)d_in[12];  // [384]
  const float* sch  = (const float*)d_in[13];  // [12]
  float* out = (float*)d_out;                  // [768,384]

  // workspace carve-up (floats; all offsets 16B aligned)
  float* ws = (float*)d_ws;
  size_t o = 0;
  float* qb     = ws + o; o += (size_t)NRES * 192;
  float* kb     = ws + o; o += (size_t)NRES * 192;
  float* vb     = ws + o; o += (size_t)NRES * 192;
  float* qpb    = ws + o; o += (size_t)NRES * 144;
  float* kpb    = ws + o; o += (size_t)NRES * 144;
  float* vpb    = ws + o; o += (size_t)NRES * 288;
  float* gq     = ws + o; o += (size_t)NRES * 144;
  float* gk     = ws + o; o += (size_t)NRES * 144;
  float* gv     = ws + o; o += (size_t)NRES * 288;
  float* sqq    = ws + o; o += (size_t)NHD * NRES;
  float* sqk    = ws + o; o += (size_t)NHD * NRES;
  float* Apack  = ws + o; o += (size_t)NHD * NRES * 32;
  float* Bpack  = ws + o; o += (size_t)NHD * NRES * 32;
  float* WbPad  = ws + o; o += (size_t)CZD * HPAD;
  float* logits = ws + o; o += (size_t)HPAD * N2;   // 16 head-rows; 12 live + 4 zero pad
  float* vgt    = ws + o; o += (size_t)NHD * 48 * NRES;
  float* ocat   = ws + o; o += (size_t)NHD * NRES * 48;
  float* att    = ws + o; o += (size_t)NRES * 2112;

  // 0) pad Wb -> [128,16]
  pad_wb<<<(CZD * HPAD + 255) / 256, 256, 0, stream>>>(Wb, WbPad);

  // 1) projections: s @ W*
  launch_gemm(stream, s, Wq,  qb,  nullptr, NRES, 192, CSD, NRES, 192, CSD, 192, 192, 1, 0, 0, 0, 0, 0);
  launch_gemm(stream, s, Wk,  kb,  nullptr, NRES, 192, CSD, NRES, 192, CSD, 192, 192, 1, 0, 0, 0, 0, 0);
  launch_gemm(stream, s, Wv,  vb,  nullptr, NRES, 192, CSD, NRES, 192, CSD, 192, 192, 1, 0, 0, 0, 0, 0);
  launch_gemm(stream, s, Wqp, qpb, nullptr, NRES, 144, CSD, NRES, 144, CSD, 144, 144, 1, 0, 0, 0, 0, 0);
  launch_gemm(stream, s, Wkp, kpb, nullptr, NRES, 144, CSD, NRES, 144, CSD, 144, 144, 1, 0, 0, 0, 0, 0);
  launch_gemm(stream, s, Wvp, vpb, nullptr, NRES, 288, CSD, NRES, 288, CSD, 288, 288, 1, 0, 0, 0, 0, 0);

  // 2) frame application + operand packing
  apply_frames<<<(NRES * 192 + 255) / 256, 256, 0, stream>>>(qpb, kpb, vpb, R, tr, gq, gk, gv);
  pack_ab<<<(NHD * NRES + 255) / 256, 256, 0, stream>>>(qb, kb, gq, gk, sch, Apack, Bpack, sqq, sqk);
  pack_vg<<<(NHD * NRES + 255) / 256, 256, 0, stream>>>(vb, gv, vgt);

  // 3) w_pair: z[N*N,128] @ WbPad[128,16], transposed store -> logits[h][i*N+j]
  //    (pad heads 12..15 receive exact zeros)
  launch_gemm(stream, z, WbPad, logits, nullptr, N2, HPAD, CZD, N2, HPAD,
              CZD, HPAD, N2, 1, 0, 0, 0, 0, 1);

  // 4) logit tiles: one K=32 WMMA on top of w_pair + gamma*(sq terms)
  {
    dim3 grid((48 * 48 + 7) / 8, NHD);
    logits_wmma<<<grid, 256, 0, stream>>>(Apack, Bpack, sqq, sqk, sch, logits);
  }

  // 5) softmax over j (in place -> attention weights 'a'; pad heads untouched = 0)
  softmax_rows<<<NHD * NRES, 256, 0, stream>>>(logits);

  // 6) per-head a[h] @ [v|gv] -> ocat[h][i][48]   (transposed-B fast path)
  launch_gemm(stream, logits, vgt, ocat, nullptr, NRES, 48, NRES, NRES, 48,
              NRES, NRES, 48, NHD, (long)N2, (long)48 * NRES, (long)NRES * 48, 1, 0);

  // 7) o_z: async-LDS staged, one workgroup per residue, straight into att
  oz_async_wmma<<<NRES, 256, 0, stream>>>(logits, z, att);

  // 8) inverse frames, pts + norms + o_s copy into att[768,2112]
  finalize_att<<<(NHD * NRES + 255) / 256, 256, 0, stream>>>(ocat, R, tr, att);

  // 9) out = att @ Wout + bout
  launch_gemm(stream, att, Wout, out, bout, NRES, CSD, 2112, NRES, CSD,
              2112, CSD, CSD, 1, 0, 0, 0, 0, 0);
}
